// Method_Citeseer_68298569941755
// MI455X (gfx1250) — matrix operand order repaired and checked by reading, not compiled, not run on hardware
//
#include <hip/hip_runtime.h>
#include <cstdint>
#include <cstddef>

// ---------------- WMMA fragment types (gfx1250, wave32) ----------------
typedef __attribute__((ext_vector_type(16))) __bf16       v16bf;
typedef __attribute__((ext_vector_type(8)))  float        v8f;
typedef __attribute__((ext_vector_type(4)))  unsigned int u32x4;

union FragBF {
  v16bf v;
  u32x4 q[2];   // two 16-byte halves: frag elements 0..7 and 8..15
};

// ---------------- cast f32 -> bf16 with column padding ----------------
__global__ void k_cast_pad_bf16(const float* __restrict__ src, __bf16* __restrict__ dst,
                                int rows, int cols, int colsPad) {
  size_t idx = (size_t)blockIdx.x * blockDim.x + threadIdx.x;
  size_t total = (size_t)rows * (size_t)colsPad;
  if (idx >= total) return;
  int    c = (int)(idx % (size_t)colsPad);
  size_t r = idx / (size_t)colsPad;
  float  f = (c < cols) ? src[r * (size_t)cols + c] : 0.0f;
  dst[idx] = (__bf16)f;
}

// ---------------- transpose + cast f32 -> bf16 with padding ----------------
// w is [K x N] row-major; wt is [Np x Kp] row-major (zero-padded)
__global__ void k_transpose_cast_pad(const float* __restrict__ w, __bf16* __restrict__ wt,
                                     int K, int N, int Kp, int Np) {
  size_t idx = (size_t)blockIdx.x * blockDim.x + threadIdx.x;
  size_t total = (size_t)Np * (size_t)Kp;
  if (idx >= total) return;
  int k = (int)(idx % (size_t)Kp);
  int n = (int)(idx / (size_t)Kp);
  float f = (n < N && k < K) ? w[(size_t)k * N + n] : 0.0f;
  wt[idx] = (__bf16)f;
}

// ---------------- bf16 WMMA GEMM:  C[M x N] = A[M x Kp] * Bt[Np x Kp]^T ----
// A row-major bf16 (Kp multiple of 32, zero padded). Bt is B transposed
// (N-major) bf16, Np >= gridDim.x*256 rows, zero padded (L2-resident).
// Block: 256 threads = 8 waves tiled 2(M) x 4(N); each wave computes a
// 16(M) x 64(N) strip with 4 f32 accumulators sharing one A fragment.
// Adjacent blockIdx.x blocks share the same A rows, so A streams from HBM
// ~once while the partner block hits L2 (A loads kept RT on purpose).
__global__ void __launch_bounds__(256)
k_gemm_bf16_wmma(const __bf16* __restrict__ A, const __bf16* __restrict__ Bt,
                 float* __restrict__ C, int M, int N, int Kp, int ldc) {
  const int lane = threadIdx.x & 31;
  const int wave = threadIdx.x >> 5;
  const int wm = wave & 1;              // 2 waves along M
  const int wn = wave >> 1;             // 4 waves along N
  const int tileM = blockIdx.y * 32 + wm * 16;
  if (tileM >= M) return;               // wave-uniform: EXEC stays all-ones
  const int n0 = blockIdx.x * 256 + wn * 64;    // wave covers n0 .. n0+63

  const int mrow = lane & 15;           // row (A) / column (B) owned by lane
  const int h    = lane >> 4;           // lane half

  const __bf16* aRow = A  + (size_t)(tileM + mrow) * Kp;
  const __bf16* bRow = Bt + (size_t)(n0   + mrow) * Kp;
  const size_t  bTileStride = (size_t)16 * Kp;   // 16 columns of B per sub-tile

  v8f acc0 = {}, acc1 = {}, acc2 = {}, acc3 = {};

  for (int k = 0; k < Kp; k += 32) {
    // A fragment (16x32 bf16): lane half h holds K = h*16 .. h*16+15
    // (elems 0..15, contiguous) -> two b128 loads.
    FragBF a;
    a.q[0] = *reinterpret_cast<const u32x4*>(aRow + k + h * 16);
    a.q[1] = *reinterpret_cast<const u32x4*>(aRow + k + h * 16 + 8);

    // Prefetch next K-step of A (gfx1250 global_prefetch_b8).
    __builtin_prefetch(aRow + k + 32 + h * 8, 0, 1);

    // B fragments (32x16 bf16): lane half h holds K = h*16 .. h*16+15,
    // contiguous in the transposed (N-major) weight -> two adjacent b128s.
    // All four loaded up-front into distinct fragments so the scheduler can
    // overlap loads with the WMMA chain (avoids wait-to-zero serialization).
    FragBF b0, b1, b2, b3;
    {
      const __bf16* p0 = bRow + 0 * bTileStride + k + h * 16;
      const __bf16* p1 = bRow + 1 * bTileStride + k + h * 16;
      const __bf16* p2 = bRow + 2 * bTileStride + k + h * 16;
      const __bf16* p3 = bRow + 3 * bTileStride + k + h * 16;
      b0.q[0] = *reinterpret_cast<const u32x4*>(p0);
      b0.q[1] = *reinterpret_cast<const u32x4*>(p0 + 8);
      b1.q[0] = *reinterpret_cast<const u32x4*>(p1);
      b1.q[1] = *reinterpret_cast<const u32x4*>(p1 + 8);
      b2.q[0] = *reinterpret_cast<const u32x4*>(p2);
      b2.q[1] = *reinterpret_cast<const u32x4*>(p2 + 8);
      b3.q[0] = *reinterpret_cast<const u32x4*>(p3);
      b3.q[1] = *reinterpret_cast<const u32x4*>(p3 + 8);
    }

    acc0 = __builtin_amdgcn_wmma_f32_16x16x32_bf16(false, a.v, false, b0.v,
                                                   (short)0, acc0, false, false);
    acc1 = __builtin_amdgcn_wmma_f32_16x16x32_bf16(false, a.v, false, b1.v,
                                                   (short)0, acc1, false, false);
    acc2 = __builtin_amdgcn_wmma_f32_16x16x32_bf16(false, a.v, false, b2.v,
                                                   (short)0, acc2, false, false);
    acc3 = __builtin_amdgcn_wmma_f32_16x16x32_bf16(false, a.v, false, b3.v,
                                                   (short)0, acc3, false, false);
  }

  // C/D layout: VGPR r of lane l holds element (M = tileM + 8*(l>>4) + r,
  //                                             N = ntile + (l&15))
  const int rowBase = tileM + h * 8;
  const int colBase = n0 + (lane & 15);
#pragma unroll
  for (int r = 0; r < 8; ++r) {
    const size_t rowOff = (size_t)(rowBase + r) * ldc;
    int c0 = colBase;
    int c1 = colBase + 16;
    int c2 = colBase + 32;
    int c3 = colBase + 48;
    if (c0 < N) C[rowOff + c0] = acc0[r];
    if (c1 < N) C[rowOff + c1] = acc1[r];
    if (c2 < N) C[rowOff + c2] = acc2[r];
    if (c3 < N) C[rowOff + c3] = acc3[r];
  }
}

// ---------------- sparse aggregation: out[dst] += w * h[src] ----------------
// One block per edge; lanes sweep the feature dimension (coalesced gather,
// dword f32 atomics -> global_atomic_add_f32). h and out are ~100MB each and
// stay L2-resident (192MB L2), so the random gathers/atomics are L2 traffic.
__global__ void k_aggregate(const float* __restrict__ h, const float* __restrict__ ew,
                            const int* __restrict__ src, const int* __restrict__ dst,
                            float* __restrict__ out, int D) {
  const int e = blockIdx.x;
  const int s = src[e];
  const int d = dst[e];
  const float w = ew[e];
  const float* hrow = h + (size_t)s * D;
  float* orow = out + (size_t)d * D;
  for (int f = threadIdx.x; f < D; f += blockDim.x) {
    atomicAdd(&orow[f], w * hrow[f]);
  }
}

// ---------------- flat aggregation for tiny D (layer 3, D=6) ---------------
__global__ void k_aggregate_flat(const float* __restrict__ h, const float* __restrict__ ew,
                                 const int* __restrict__ src, const int* __restrict__ dst,
                                 float* __restrict__ out, int E, int D) {
  size_t idx = (size_t)blockIdx.x * blockDim.x + threadIdx.x;
  size_t total = (size_t)E * (size_t)D;
  if (idx >= total) return;
  int e = (int)(idx / (size_t)D);
  int f = (int)(idx % (size_t)D);
  int s = src[e];
  int d = dst[e];
  atomicAdd(&out[(size_t)d * D + f], ew[e] * h[(size_t)s * D + f]);
}

// ---------------- bias + relu, in place ----------------
__global__ void k_bias_relu(float* __restrict__ x, const float* __restrict__ b,
                            size_t total, int D) {
  size_t idx = (size_t)blockIdx.x * blockDim.x + threadIdx.x;
  if (idx >= total) return;
  int d = (int)(idx % (size_t)D);
  float v = x[idx] + b[d];
  x[idx] = fmaxf(v, 0.0f);
}

// ---------------- bias + row-wise log_softmax (C <= 8) ----------------
__global__ void k_bias_logsoftmax(const float* __restrict__ in, const float* __restrict__ b,
                                  float* __restrict__ out, int M, int C) {
  int r = blockIdx.x * blockDim.x + threadIdx.x;
  if (r >= M) return;
  float v[8];
  float mx = -INFINITY;
  const float* row = in + (size_t)r * C;
  for (int c = 0; c < C; ++c) { v[c] = row[c] + b[c]; mx = fmaxf(mx, v[c]); }
  float s = 0.0f;
  for (int c = 0; c < C; ++c) s += expf(v[c] - mx);
  float lse = mx + logf(s);
  float* orow = out + (size_t)r * C;
  for (int c = 0; c < C; ++c) orow[c] = v[c] - lse;
}

// ======================= host-side orchestration =======================
static inline size_t cdiv_sz(size_t a, size_t b) { return (a + b - 1) / b; }
static inline int    cdiv_i(int a, int b)        { return (a + b - 1) / b; }
static inline int    round_up(int a, int b)      { return cdiv_i(a, b) * b; }

extern "C" void kernel_launch(void* const* d_in, const int* in_sizes, int n_in,
                              void* d_out, int out_size, void* d_ws, size_t ws_size,
                              hipStream_t stream) {
  // inputs: x, w1, b1, w2, b2, w3, b3, edge_w, edge_src, edge_dst
  const float* x   = (const float*)d_in[0];
  const float* w1  = (const float*)d_in[1];
  const float* b1  = (const float*)d_in[2];
  const float* w2  = (const float*)d_in[3];
  const float* b2  = (const float*)d_in[4];
  const float* w3  = (const float*)d_in[5];
  const float* b3  = (const float*)d_in[6];
  const float* ew  = (const float*)d_in[7];
  const int*   esr = (const int*)d_in[8];
  const int*   eds = (const int*)d_in[9];
  float* out = (float*)d_out;

  const int HID = in_sizes[2];                  // 500
  const int IN  = in_sizes[1] / HID;            // 3703
  const int CLS = in_sizes[6];                  // 6
  const int M   = in_sizes[0] / IN;             // 50000
  const int E   = in_sizes[7];                  // 800000

  const int Kp1 = round_up(IN, 32);             // 3712
  const int Kp2 = round_up(HID, 32);            // 512
  const int NP  = 512;                          // B rows touched (grid.x*256)

  // ---- workspace carve-out (256B aligned) ----
  char* ws = (char*)d_ws;
  auto carve = [&](size_t bytes) {
    char* p = ws;
    ws += (bytes + 255) & ~(size_t)255;
    return p;
  };
  __bf16* Xb  = (__bf16*)carve((size_t)M * Kp1 * sizeof(__bf16));   // padded bf16 X
  __bf16* Hb  = (__bf16*)carve((size_t)M * Kp2 * sizeof(__bf16));   // padded bf16 H
  float*  P   = (float*)carve((size_t)M * HID * sizeof(float));     // GEMM output
  float*  Cg  = (float*)carve((size_t)M * HID * sizeof(float));     // aggregation out
  __bf16* W1t = (__bf16*)carve((size_t)NP * Kp1 * sizeof(__bf16));
  __bf16* W2t = (__bf16*)carve((size_t)NP * Kp2 * sizeof(__bf16));
  __bf16* W3t = (__bf16*)carve((size_t)NP * Kp2 * sizeof(__bf16));
  (void)ws_size;

  const int T = 256;

  // ===================== layer 1 =====================
  {
    size_t tot = (size_t)M * Kp1;
    k_cast_pad_bf16<<<dim3((unsigned)cdiv_sz(tot, T)), T, 0, stream>>>(x, Xb, M, IN, Kp1);
  }
  {
    size_t tot = (size_t)NP * Kp1;
    k_transpose_cast_pad<<<dim3((unsigned)cdiv_sz(tot, T)), T, 0, stream>>>(w1, W1t, IN, HID, Kp1, NP);
  }
  {
    dim3 grid((unsigned)cdiv_i(HID, 256), (unsigned)cdiv_i(M, 32));
    k_gemm_bf16_wmma<<<grid, T, 0, stream>>>(Xb, W1t, P, M, HID, Kp1, HID);
  }
  hipMemsetAsync(Cg, 0, (size_t)M * HID * sizeof(float), stream);
  k_aggregate<<<dim3((unsigned)E), T, 0, stream>>>(P, ew, esr, eds, Cg, HID);
  {
    size_t tot = (size_t)M * HID;
    k_bias_relu<<<dim3((unsigned)cdiv_sz(tot, T)), T, 0, stream>>>(Cg, b1, tot, HID);
  }

  // ===================== layer 2 =====================
  {
    size_t tot = (size_t)M * Kp2;
    k_cast_pad_bf16<<<dim3((unsigned)cdiv_sz(tot, T)), T, 0, stream>>>(Cg, Hb, M, HID, Kp2);
  }
  {
    size_t tot = (size_t)NP * Kp2;
    k_transpose_cast_pad<<<dim3((unsigned)cdiv_sz(tot, T)), T, 0, stream>>>(w2, W2t, HID, HID, Kp2, NP);
  }
  {
    dim3 grid((unsigned)cdiv_i(HID, 256), (unsigned)cdiv_i(M, 32));
    k_gemm_bf16_wmma<<<grid, T, 0, stream>>>(Hb, W2t, P, M, HID, Kp2, HID);
  }
  hipMemsetAsync(Cg, 0, (size_t)M * HID * sizeof(float), stream);
  k_aggregate<<<dim3((unsigned)E), T, 0, stream>>>(P, ew, esr, eds, Cg, HID);
  {
    size_t tot = (size_t)M * HID;
    k_bias_relu<<<dim3((unsigned)cdiv_sz(tot, T)), T, 0, stream>>>(Cg, b2, tot, HID);
  }

  // ===================== layer 3 =====================
  {
    size_t tot = (size_t)M * Kp2;
    k_cast_pad_bf16<<<dim3((unsigned)cdiv_sz(tot, T)), T, 0, stream>>>(Cg, Hb, M, HID, Kp2);
  }
  {
    size_t tot = (size_t)NP * Kp2;
    k_transpose_cast_pad<<<dim3((unsigned)cdiv_sz(tot, T)), T, 0, stream>>>(w3, W3t, HID, CLS, Kp2, NP);
  }
  {
    dim3 grid(1u, (unsigned)cdiv_i(M, 32));
    k_gemm_bf16_wmma<<<grid, T, 0, stream>>>(Hb, W3t, P, M, CLS, Kp2, CLS);
  }
  hipMemsetAsync(Cg, 0, (size_t)M * CLS * sizeof(float), stream);
  {
    size_t tot = (size_t)E * CLS;
    k_aggregate_flat<<<dim3((unsigned)cdiv_sz(tot, T)), T, 0, stream>>>(P, ew, esr, eds, Cg, E, CLS);
  }
  k_bias_logsoftmax<<<dim3((unsigned)cdiv_i(M, T)), T, 0, stream>>>(Cg, b3, out, M, CLS);
}
